// LiftSplatSimple_63874753626195
// MI455X (gfx1250) — matrix operand order repaired and checked by reading, not compiled, not run on hardware
//
#include <hip/hip_runtime.h>
#include <hip/hip_bf16.h>
#include <math.h>

// ---------------------------------------------------------------------------
// MI455X (gfx1250) Lift-Splat pipeline.
// Convs = implicit-GEMM on v_wmma_f32_16x16x32_f16 (wave32), BN folded,
// f16 NHWC activations, f32 accumulation. Weight tiles are DMA'd into LDS
// by the Tensor Data Mover (with LDS row padding done by TDM pad_enable);
// cooperative vector loads are the fallback when the builtin is absent.
// ---------------------------------------------------------------------------

typedef __attribute__((ext_vector_type(16))) _Float16 v16h;
typedef __attribute__((ext_vector_type(8)))  _Float16 v8h;
typedef __attribute__((ext_vector_type(8)))  float    v8f;
typedef __attribute__((ext_vector_type(4)))  unsigned int v4u;
typedef __attribute__((ext_vector_type(8)))  int      v8i;
typedef __attribute__((ext_vector_type(4)))  int      v4i;

#define BN_EPS 1e-5f

#if __has_builtin(__builtin_amdgcn_tensor_load_to_lds) && \
    __has_builtin(__builtin_amdgcn_s_wait_tensorcnt)
#define USE_TDM 1
#else
#define USE_TDM 0
#endif

// ---------------- weight packing (fold BN scale) ---------------------------
// Packed layout: [Cout][K] with K = (kh*KW+kw)*Cin + c  (K-contiguous rows,
// matching both the TDM 2D tile and the WMMA B-fragment gather).
__global__ void pack_weight_kernel(const float* __restrict__ w,
                                   const float* __restrict__ g,
                                   _Float16* __restrict__ dst,
                                   int Cin, int Cout, int KH, int KW) {
  int idx = blockIdx.x * 256 + threadIdx.x;
  int K = Cin * KH * KW;
  if (idx >= K * Cout) return;
  int co = idx / K, k = idx - co * K;
  int patch = k / Cin, c = k - patch * Cin;
  int kh = patch / KW, kw = patch - kh * KW;
  float s = g ? g[co] * rsqrtf(1.0f + BN_EPS) : 1.0f;
  dst[idx] = (_Float16)(w[(((size_t)co * Cin + c) * KH + kh) * KW + kw] * s);
}

__global__ void fold_bias_kernel(const float* __restrict__ b,
                                 const float* __restrict__ g,
                                 const float* __restrict__ be,
                                 float* __restrict__ dst, int Cout) {
  int i = blockIdx.x * 256 + threadIdx.x;
  if (i >= Cout) return;
  float s = g ? g[i] * rsqrtf(1.0f + BN_EPS) : 1.0f;
  dst[i] = b[i] * s + (be ? be[i] : 0.0f);
}

// ---------------- layout conversions ---------------------------------------
__global__ void nchw_to_nhwc_h_kernel(const float* __restrict__ s,
                                      _Float16* __restrict__ d,
                                      int C, int HW, int total) {
  int i = blockIdx.x * 256 + threadIdx.x;
  if (i >= total) return;
  int c = i % C; int t = i / C; int yx = t % HW; int b = t / HW;
  d[i] = (_Float16)s[((size_t)b * C + c) * HW + yx];
}

__global__ void nhwc_h_to_nchw_f_kernel(const _Float16* __restrict__ s,
                                        float* __restrict__ d,
                                        int C, int HW, int total) {
  int i = blockIdx.x * 256 + threadIdx.x;
  if (i >= total) return;
  int yx = i % HW; int t = i / HW; int c = t % C; int b = t / C;
  d[i] = (float)s[((size_t)b * HW + yx) * C + c];
}

__global__ void f32_to_h_kernel(const float* __restrict__ s,
                                _Float16* __restrict__ d, int n) {
  int i = blockIdx.x * 256 + threadIdx.x;
  if (i < n) d[i] = (_Float16)s[i];
}

// ---------------- implicit-GEMM conv on WMMA --------------------------------
// Block: 256 threads = 8 waves. Tile: 64 pixels (M) x 128 Cout (N).
// Wave (wm 0..3, wn 0..1): 16 pixels x 64 couts = 4 accumulators.
// K walked in 32-steps. Requires Cin % 32 == 0 and nPix % 64 == 0.
__global__ __launch_bounds__(256)
void conv_wmma_kernel(const _Float16* __restrict__ src,   // NHWC [B*HW][Cin]
                      const _Float16* __restrict__ wgt,   // [Cout][K]
                      const float*    __restrict__ bias,  // [Cout]
                      const _Float16* __restrict__ res,   // NHWC [B*HW][Cout] or null
                      _Float16*       __restrict__ dst,   // NHWC [B*HW][Cout]
                      int H, int W, int Cin, int Cout,
                      int KH, int KW, int pad, int relu) {
  __shared__ alignas(16) _Float16 sA[64][40];     // 64 pixels x 32 k (+8 pad)
  __shared__ alignas(16) _Float16 sBt[128][40];   // 128 cout x 32 k (+8 pad)

  const int tid  = threadIdx.x;
  const int lane = tid & 31;
  const int wave = tid >> 5;
  const int wm   = wave & 3;      // pixel group
  const int wn   = wave >> 2;     // cout group
  const int hi   = lane >> 4;
  const int ln   = lane & 15;

  const int pixBase  = blockIdx.x * 64;
  const int coutBase = blockIdx.y * 128;
  const int HW = H * W;
  const int K  = Cin * KH * KW;

  v8f acc0 = {0, 0, 0, 0, 0, 0, 0, 0};
  v8f acc[4] = {acc0, acc0, acc0, acc0};

  // A-tile loader coords: 4 threads per pixel row, 8 halves each.
  const int aRow = tid >> 2;
  const int aSeg = (tid & 3) * 8;
  const int p    = pixBase + aRow;
  const int bImg = p / HW;
  const int rem  = p - bImg * HW;
  const int py   = rem / W;
  const int px   = rem - py * W;

#if !USE_TDM
  // Fallback B loader: 2 threads per cout row, 16 halves each.
  const int bCol = tid >> 1;
  const int bSeg = (tid & 1) * 16;
#endif

#if USE_TDM
  // ---- Tensor DMA descriptor (D#), built once; global_addr updated per step.
  // 2D tile: dim0 = K (contiguous), dim1 = cout rows. tile = 32 x 128.
  // pad_enable inserts 4 DWORDs after every 16 DWORDs -> 40-half LDS rows.
  unsigned int ldsAddr = 0;
  unsigned long long gaBase = 0;
  int tdim1 = Cout - coutBase;          // remaining rows; OOB rows -> zeros
  if (wave == 0) {
    ldsAddr = (unsigned int)(unsigned long long)(uintptr_t)
        (__attribute__((address_space(3))) _Float16*)&sBt[0][0];
    gaBase = (unsigned long long)(uintptr_t)(wgt + (size_t)coutBase * K);
  }
  v4u g0;
  v8i g1;
  v4i gz4;
  v8i gz8;
  gz4[0] = 0; gz4[1] = 0; gz4[2] = 0; gz4[3] = 0;
#pragma unroll
  for (int q = 0; q < 8; ++q) gz8[q] = 0;
  g0[0] = 1u;                                   // count=1, user mode
  g0[1] = ldsAddr;                              // lds_addr (bytes)
  g0[2] = 0;                                    // global_addr lo (per step)
  g0[3] = (2u << 30);                           // type=2 ("image")
  g1[0] = (1 << 16) |                           // data_size = 2 bytes
          (1 << 20) |                           // pad_enable
          (3 << 22) |                           // pad_interval = 16 DWORDs
          (3 << 25);                            // pad_amount   = 4 DWORDs
  g1[1] = (K & 0xffff) << 16;                   // tensor_dim0[15:0]
  g1[2] = ((K >> 16) & 0xffff) | ((tdim1 & 0xffff) << 16);
  g1[3] = ((tdim1 >> 16) & 0xffff) | (32 << 16); // tile_dim0 = 32
  g1[4] = 128;                                   // tile_dim1 = 128, tile_dim2=0
  g1[5] = K;                                     // tensor_dim0_stride lo32
  g1[6] = 0;
  g1[7] = 0;
#endif

  for (int k0 = 0; k0 < K; k0 += 32) {
    // ---- stage A (im2col row fragment) into LDS
    {
      int kk    = k0 + aSeg;
      int patch = kk / Cin;
      int c     = kk - patch * Cin;
      int kh    = patch / KW, kw = patch - kh * KW;
      int sy    = py + kh - pad, sx = px + kw - pad;
      v8h av;
#pragma unroll
      for (int q = 0; q < 8; ++q) av[q] = (_Float16)0.0f;
      if (sy >= 0 && sy < H && sx >= 0 && sx < W)
        av = *(const v8h*)(src + ((size_t)(bImg * HW + sy * W + sx)) * Cin + c);
      *(v8h*)(&sA[aRow][aSeg]) = av;
    }

    // ---- stage B (weight tile, K-contiguous rows) into LDS
#if USE_TDM
    if (wave == 0) {
      unsigned long long ga = gaBase + (unsigned long long)k0 * 2u;
      g0[2] = (unsigned int)(ga & 0xffffffffu);
      g0[3] = (unsigned int)((ga >> 32) & 0x01ffffffu) | (2u << 30);
      __builtin_amdgcn_tensor_load_to_lds(g0, g1, gz4, gz4, gz8, 0);
      __builtin_amdgcn_s_wait_tensorcnt(0);
    }
#else
    {
      const int co = coutBase + bCol;
      if (co < Cout) {
        const _Float16* wp = wgt + (size_t)co * K + k0 + bSeg;
        *(v8h*)(&sBt[bCol][bSeg])     = *(const v8h*)wp;
        *(v8h*)(&sBt[bCol][bSeg + 8]) = *(const v8h*)(wp + 8);
      } else {
        v8h z;
#pragma unroll
        for (int q = 0; q < 8; ++q) z[q] = (_Float16)0.0f;
        *(v8h*)(&sBt[bCol][bSeg])     = z;
        *(v8h*)(&sBt[bCol][bSeg + 8]) = z;
      }
    }
    if (k0 + 32 < K)   // prefetch next weight slab (global_prefetch_b8)
      __builtin_prefetch(wgt + (size_t)(coutBase + bCol) * K + k0 + 32, 0, 1);
#endif

    __syncthreads();

    // ---- gather fragments (contiguous 16B LDS vector loads) and issue WMMA
    const int arow = wm * 16 + ln;
    v8h alo = *(const v8h*)(&sA[arow][hi * 8]);        // k = hi*8 .. +7
    v8h ahi = *(const v8h*)(&sA[arow][16 + hi * 8]);   // k = 16+hi*8 .. +7
    v16h aFrag = __builtin_shufflevector(alo, ahi, 0, 1, 2, 3, 4, 5, 6, 7,
                                         8, 9, 10, 11, 12, 13, 14, 15);
#pragma unroll
    for (int f = 0; f < 4; ++f) {
      const int col = wn * 64 + f * 16 + ln;
      v8h blo = *(const v8h*)(&sBt[col][hi * 8]);      // k = hi*8 .. +7
      v8h bhi = *(const v8h*)(&sBt[col][16 + hi * 8]); // k = 16+hi*8 .. +7
      v16h bFrag = __builtin_shufflevector(blo, bhi, 0, 1, 2, 3, 4, 5, 6, 7,
                                           8, 9, 10, 11, 12, 13, 14, 15);
      acc[f] = __builtin_amdgcn_wmma_f32_16x16x32_f16(
          false, aFrag, false, bFrag, (short)0, acc[f], false, false);
    }
    __syncthreads();
  }

  // ---- epilogue: bias + ReLU + residual-after-activation, store f16 NHWC
#pragma unroll
  for (int f = 0; f < 4; ++f) {
    const int co = coutBase + wn * 64 + f * 16 + ln;
    if (co < Cout) {
      const float bv = bias[co];
#pragma unroll
      for (int r = 0; r < 8; ++r) {
        const int prow = pixBase + wm * 16 + hi * 8 + r;
        float v = acc[f][r] + bv;
        if (relu) v = v > 0.0f ? v : 0.0f;
        if (res)  v += (float)res[(size_t)prow * Cout + co];
        dst[(size_t)prow * Cout + co] = (_Float16)v;
      }
    }
  }
}

// ---------------- depth softmax (D=48), NHWC-f16 logits -> NCHW-f32 --------
__global__ void softmax_depth_kernel(const _Float16* __restrict__ logits,
                                     float* __restrict__ out) {
  int p = blockIdx.x * 256 + threadIdx.x;      // b*4096 + yx
  if (p >= 8 * 4096) return;
  const _Float16* lp = logits + (size_t)p * 48;
  float v[48];
  float m = -1e30f;
#pragma unroll
  for (int d = 0; d < 48; ++d) { v[d] = (float)lp[d]; m = fmaxf(m, v[d]); }
  float s = 0.0f;
#pragma unroll
  for (int d = 0; d < 48; ++d) { v[d] = expf(v[d] - m); s += v[d]; }
  const float inv = 1.0f / s;
  const int b = p >> 12, yx = p & 4095;
#pragma unroll
  for (int d = 0; d < 48; ++d)
    out[((size_t)(b * 48 + d) << 12) + yx] = v[d] * inv;
}

// ---------------- 1-channel 1x1 heads (Cin=64), optional sigmoid*48+2 ------
__global__ void head1_kernel(const _Float16* __restrict__ src,
                             const float* __restrict__ w,
                             const float* __restrict__ b,
                             float* __restrict__ out, int nPix, int mode) {
  int p = blockIdx.x * 256 + threadIdx.x;
  if (p >= nPix) return;
  float s = b[0];
  const _Float16* sp = src + (size_t)p * 64;
#pragma unroll
  for (int c = 0; c < 64; ++c) s += (float)sp[c] * w[c];
  if (mode) s = 1.0f / (1.0f + expf(-s)) * 48.0f + 2.0f;
  out[p] = s;
}

// ---------------- lift + splat ---------------------------------------------
// BEV cell depends only on (d, x-column): reduce over the 64 image rows per
// (b,d,x) with lane=channel, then one f32 atomic scatter per channel.
__global__ __launch_bounds__(256)
void lift_splat_kernel(const _Float16* __restrict__ ctx,   // [B][64][64][32]
                       const float*    __restrict__ dp,    // [B][48][64][64]
                       float*          __restrict__ bev) { // [B][10000][32]
  const int gw   = blockIdx.x * 8 + (threadIdx.x >> 5);    // 0 .. 8*48*64-1
  const int lane = threadIdx.x & 31;                       // channel
  const int x = gw & 63;
  const int t = gw >> 6;
  const int d = t % 48;
  const int b = t / 48;

  const float depth = 2.0f + (float)d * (48.0f / 47.0f);
  const float rayx  = ((float)x * 16.0f + 8.0f - 512.0f) / 512.0f;
  const float x3    = rayx * depth;
  const int   xi    = (int)((x3 + 25.0f) * 2.0f);   // trunc toward 0
  const int   yi    = (int)(depth * 2.0f);
  const bool valid = (x3 + 25.0f >= 0.0f) && (xi >= 0) && (xi < 100) &&
                     (yi >= 0) && (yi < 100);

  float sum = 0.0f;
  const _Float16* cp  = ctx + ((size_t)b * 4096 + x) * 32 + lane;  // +y*64*32
  const float*    dpp = dp + ((size_t)(b * 48 + d) * 64) * 64 + x; // +y*64
#pragma unroll 4
  for (int y = 0; y < 64; ++y)
    sum += (float)cp[(size_t)y * 64 * 32] * dpp[(size_t)y * 64];

  if (valid)
    atomicAdd(&bev[((size_t)b * 10000 + yi * 100 + xi) * 32 + lane], sum);
}

// ===========================================================================
// Host orchestration
// ===========================================================================
struct CD { int wi, bi, gi, bei, Cin, Cout, KH, pad; };

static void run_conv(const _Float16* src, const _Float16* w, const float* bias,
                     const _Float16* res, _Float16* dst, int nPix, int H, int W,
                     int Cin, int Cout, int KH, int pad, int relu,
                     hipStream_t s) {
  dim3 g(nPix / 64, (Cout + 127) / 128);
  conv_wmma_kernel<<<g, 256, 0, s>>>(src, w, bias, res, dst, H, W, Cin, Cout,
                                     KH, KH, pad, relu);
}

extern "C" void kernel_launch(void* const* d_in, const int* in_sizes, int n_in,
                              void* d_out, int out_size, void* d_ws, size_t ws_size,
                              hipStream_t stream) {
  const float* features = (const float*)d_in[0];
  auto P = [&](int i) { return (const float*)d_in[i]; };

  // ---- workspace pools (lifetimes scheduled so reuse is safe) -------------
  const size_t POOL = 22u * 1024u * 1024u;
  char* ws = (char*)d_ws;
  char* P0 = ws;                 // featH -> dr1 -> e1
  char* P1 = ws + POOL;          // h1 -> ctx1 -> dd -> bev_raw -> e2 -> rd/vh
  char* P2 = ws + 2 * POOL;      // skip -> ctx(+bevHalf) -> bev_features
  char* P3 = ws + 3 * POOL;      // h -> bev_in
  char* WREG = ws + 4 * POOL;    // packed f16 weights (1,943,552 halves)
  char* BREG = WREG + 4u * 1024u * 1024u;   // folded biases

  // ---- conv table (indices into d_in; -1 => none) -------------------------
  static const CD cv[13] = {
    { 9,10,-1,-1, 256,256,1,0},  // 0 skip
    { 1, 2, 3, 4, 256,256,3,1},  // 1 db1
    { 5, 6, 7, 8, 256,256,3,1},  // 2 db2
    {17,18,19,20, 256,128,3,1},  // 3 ctx1
    {21,22,-1,-1, 128, 32,1,0},  // 4 ctx2
    {11,12,-1,-1, 256, 48,1,0},  // 5 dd
    {13,14,-1,-1, 256, 64,1,0},  // 6 dr1
    {23,24,-1,-1,  32, 64,1,0},  // 7 bc
    {25,26,27,28,  64,128,3,1},  // 8 be1
    {29,30,31,32, 128,128,3,1},  // 9 be2
    {33,34,35,36, 128, 64,3,1},  // 10 be3
    {37,38,-1,-1,  64, 64,3,1},  // 11 rd1
    {41,42,-1,-1,  64, 64,3,1},  // 12 vh1
  };

  _Float16* wptr[13];
  float*    bptr[13];
  {
    size_t woff = 0, boff = 0;
    for (int i = 0; i < 13; ++i) {
      const CD& c = cv[i];
      int K = c.Cin * c.KH * c.KH;
      wptr[i] = (_Float16*)WREG + woff;
      bptr[i] = (float*)BREG + boff;
      woff += (size_t)K * c.Cout;
      boff += (size_t)c.Cout;
      int tot = K * c.Cout;
      pack_weight_kernel<<<(tot + 255) / 256, 256, 0, stream>>>(
          P(c.wi), c.gi >= 0 ? P(c.gi) : nullptr, wptr[i],
          c.Cin, c.Cout, c.KH, c.KH);
      fold_bias_kernel<<<(c.Cout + 255) / 256, 256, 0, stream>>>(
          P(c.bi), c.gi >= 0 ? P(c.gi) : nullptr,
          c.bei >= 0 ? P(c.bei) : nullptr, bptr[i], c.Cout);
    }
  }

  // ---- buffers ------------------------------------------------------------
  _Float16* featH  = (_Float16*)P0;               // [8*4096][256]
  _Float16* h1     = (_Float16*)P1;               // [8*4096][256]
  _Float16* skipO  = (_Float16*)P2;               // [8*4096][256]
  _Float16* hbuf   = (_Float16*)P3;               // [8*4096][256]
  _Float16* ctx1   = (_Float16*)P1;               // [8*4096][128]
  _Float16* ctxh   = (_Float16*)P2;               // [8*4096][32]   (P2+0)
  _Float16* ddlog  = (_Float16*)P1;               // [8*4096][48]
  _Float16* dr1b   = (_Float16*)P0;               // [8*4096][64]
  float*    bevRaw = (float*)P1;                  // [8][10000][32] f32
  _Float16* bevH   = (_Float16*)(P2 + 8u*1024u*1024u); // [8*10000][32]
  _Float16* bevIn  = (_Float16*)P3;               // [8*10000][64]
  _Float16* e1     = (_Float16*)P0;               // [8*10000][128]
  _Float16* e2     = (_Float16*)P1;               // [8*10000][128]
  _Float16* bevF   = (_Float16*)P2;               // [8*10000][64]
  _Float16* headT  = (_Float16*)P1;               // [8*10000][64]

  float* outF   = (float*)d_out;
  const int N_BEVF = 8 * 64 * 100 * 100;          // 5,120,000
  const int N_ROAD = 8 * 100 * 100;               // 80,000
  float* outBev  = outF;
  float* outRoad = outF + N_BEVF;
  float* outVeh  = outRoad + N_ROAD;
  float* outDP   = outVeh + N_ROAD;               // [8][48][64][64]
  float* outDR   = outDP + 8 * 48 * 64 * 64;      // [8][1][64][64]

  const int PIX_F = 8 * 64 * 64;                  // 32768 feature pixels
  const int PIX_B = 8 * 100 * 100;                // 80000 BEV pixels

  // ---- features NCHW f32 -> NHWC f16 --------------------------------------
  {
    int tot = PIX_F * 256;
    nchw_to_nhwc_h_kernel<<<(tot + 255) / 256, 256, 0, stream>>>(
        features, featH, 256, 4096, tot);
  }

  // ---- depth trunk --------------------------------------------------------
  run_conv(featH, wptr[0], bptr[0], nullptr, skipO, PIX_F, 64, 64, 256, 256, 1, 0, 0, stream); // skip
  run_conv(featH, wptr[1], bptr[1], nullptr, h1,    PIX_F, 64, 64, 256, 256, 3, 1, 1, stream); // db1
  run_conv(h1,    wptr[2], bptr[2], skipO,   hbuf,  PIX_F, 64, 64, 256, 256, 3, 1, 1, stream); // db2 + skip

  // ---- context ------------------------------------------------------------
  run_conv(featH, wptr[3], bptr[3], nullptr, ctx1, PIX_F, 64, 64, 256, 128, 3, 1, 1, stream);  // ctx1
  run_conv(ctx1,  wptr[4], bptr[4], nullptr, ctxh, PIX_F, 64, 64, 128,  32, 1, 0, 0, stream);  // ctx2

  // ---- depth heads --------------------------------------------------------
  run_conv(hbuf, wptr[5], bptr[5], nullptr, ddlog, PIX_F, 64, 64, 256, 48, 1, 0, 0, stream);   // dd
  run_conv(hbuf, wptr[6], bptr[6], nullptr, dr1b,  PIX_F, 64, 64, 256, 64, 1, 0, 1, stream);   // dr1

  softmax_depth_kernel<<<(PIX_F + 255) / 256, 256, 0, stream>>>(ddlog, outDP);
  head1_kernel<<<(PIX_F + 255) / 256, 256, 0, stream>>>(dr1b, P(15), P(16),
                                                        outDR, PIX_F, 1);

  // ---- lift + splat -------------------------------------------------------
  (void)hipMemsetAsync(bevRaw, 0, (size_t)8 * 10000 * 32 * sizeof(float), stream);
  lift_splat_kernel<<<(8 * 48 * 64) / 8, 256, 0, stream>>>(ctxh, outDP, bevRaw);
  {
    int n = 8 * 10000 * 32;
    f32_to_h_kernel<<<(n + 255) / 256, 256, 0, stream>>>(bevRaw, bevH, n);
  }

  // ---- BEV encoder --------------------------------------------------------
  run_conv(bevH,  wptr[7],  bptr[7],  nullptr, bevIn, PIX_B, 100, 100,  32,  64, 1, 0, 0, stream); // bc
  run_conv(bevIn, wptr[8],  bptr[8],  nullptr, e1,    PIX_B, 100, 100,  64, 128, 3, 1, 1, stream); // be1
  run_conv(e1,    wptr[9],  bptr[9],  nullptr, e2,    PIX_B, 100, 100, 128, 128, 3, 1, 1, stream); // be2
  run_conv(e2,    wptr[10], bptr[10], bevIn,   bevF,  PIX_B, 100, 100, 128,  64, 3, 1, 1, stream); // be3 + res

  {
    int tot = PIX_B * 64;
    nhwc_h_to_nchw_f_kernel<<<(tot + 255) / 256, 256, 0, stream>>>(
        bevF, outBev, 64, 10000, tot);
  }

  // ---- road / vehicle heads ----------------------------------------------
  run_conv(bevF, wptr[11], bptr[11], nullptr, headT, PIX_B, 100, 100, 64, 64, 3, 1, 1, stream); // rd1
  head1_kernel<<<(PIX_B + 255) / 256, 256, 0, stream>>>(headT, P(39), P(40),
                                                        outRoad, PIX_B, 0);
  run_conv(bevF, wptr[12], bptr[12], nullptr, headT, PIX_B, 100, 100, 64, 64, 3, 1, 1, stream); // vh1
  head1_kernel<<<(PIX_B + 255) / 256, 256, 0, stream>>>(headT, P(43), P(44),
                                                        outVeh, PIX_B, 0);
}
